// SRNet_34110630265631
// MI455X (gfx1250) — matrix operand order repaired and checked by reading, not compiled, not run on hardware
//
#include <hip/hip_runtime.h>
#include <hip/hip_bf16.h>

// ---------------------------------------------------------------------------
// MI455X (gfx1250) implementation of the SR text-prior network.
// Core op: implicit-GEMM conv via v_wmma_f32_16x16x32_bf16 (wave32).
//   - Block = 128 threads (4 waves); block tile = 64 pixels x 64 couts.
//   - K contracted 64 at a time through a DOUBLE-BUFFERED LDS pipeline:
//     stage slab k+1 while slab k feeds the matrix pipes.
//   - Each wave: 16-pixel slice x 4 cout tiles x 2 K-halves = 8 WMMA per slab,
//     A operand registers built once and reused across the 4 cout tiles.
//   - Templated on (ks, stride); spatial dims are powers of two -> shift/mask
//     addressing; per-k im2col decode hoisted to once per thread per slab.
//   - Epilogue: LDS transpose for pixel-contiguous coalesced stores with
//     fused bias + lrelu/tanh + per-image column mask.
// ---------------------------------------------------------------------------

typedef __bf16 bf16;
typedef __attribute__((ext_vector_type(16))) __bf16 v16bf;
typedef __attribute__((ext_vector_type(8)))  __bf16 v8bf;
typedef __attribute__((ext_vector_type(8)))  float  v8f;

static __device__ __forceinline__ int iclampi(int v, int lo, int hi) {
    return v < lo ? lo : (v > hi ? hi : v);
}

#define AST2 72            // LDS row stride in bf16 (144 B): 16B-aligned, bank-spread
#define ABUF (64 * AST2)   // one K-slab buffer (64 rows x 64 kk, padded)

template<int KS, int STRIDE>
__global__ void __launch_bounds__(128)
k_conv_wmma(const bf16* __restrict__ X, const bf16* __restrict__ Wt,
            const float* __restrict__ Bias, bf16* __restrict__ Y,
            int N, int Cin, int H, int W, int Cout, int Ho, int WoShift,
            int act, const int* __restrict__ wv)
{
    constexpr int KK  = KS * KS;
    constexpr int PAD = (KS - 1) / 2;

    __shared__ bf16  As[2 * ABUF];   // [buf][pixel m][kk]
    __shared__ bf16  Bs[2 * ABUF];   // [buf][cout_local][kk]
    __shared__ float Cs[64 * 65];    // [cout_local][pixel m], padded

    const int tid   = threadIdx.x;
    const int wave  = tid >> 5;
    const int lane  = tid & 31;
    const int half  = lane >> 4;
    const int ncol  = lane & 15;
    const int Wo    = 1 << WoShift;
    const int HWo   = Ho << WoShift;
    const int pbase = blockIdx.x * 64;
    const int cbase = blockIdx.y * 64;
    const int n     = blockIdx.z;
    const int K     = Cin * KK;
    const long xbase = (long)n * Cin * H * W;

    v8f acc[4] = {};

    // ---- stage one 64-wide K-slab into buffer `buf` ----
    // thread -> fixed kk = tid&63 (k decode hoisted), rows m = (tid>>6) + 2*it
    auto stage = [&](int buf, int k0) {
        const int kk  = tid & 63;
        const int k   = k0 + kk;
        const bool kok = (k < K);
        const int cin = k / KK;
        const int r   = k - cin * KK;
        const int kh  = r / KS;
        const int kw  = r - kh * KS;
        const long xcin = xbase + (long)cin * H * W;
        const int mb  = tid >> 6;          // 0 or 1
        bf16* Ab = As + buf * ABUF;
        bf16* Bb = Bs + buf * ABUF;
#pragma unroll
        for (int it = 0; it < 32; ++it) {
            int m = mb + it * 2;
            int p = pbase + m;
            float v = 0.f;
            if (kok && p < HWo) {
                int oy = p >> WoShift, ox = p & (Wo - 1);
                int iy = oy * STRIDE + kh - PAD;
                int ix = ox * STRIDE + kw - PAD;
                if (iy >= 0 && iy < H && ix >= 0 && ix < W)
                    v = (float)X[xcin + (long)iy * W + ix];
            }
            Ab[m * AST2 + kk] = (bf16)v;
        }
#pragma unroll
        for (int it = 0; it < 32; ++it) {
            int cl   = mb + it * 2;
            int cout = cbase + cl;
            float v = 0.f;
            if (kok && cout < Cout)
                v = (float)Wt[(long)cout * K + k];   // [Cout][Cin*KS*KS]
            Bb[cl * AST2 + kk] = (bf16)v;
        }
    };

    stage(0, 0);

    for (int k0 = 0; k0 < K; k0 += 64) {
        const int cur = (k0 >> 6) & 1;
        __syncthreads();                       // slab `cur` ready
        if (k0 + 64 < K) {
            // CDNA5 L2 prefetch of the next slab (global_prefetch_b8)
            __builtin_prefetch(Wt + (long)cbase * K + k0 + 64 + tid, 0, 3);
            stage(cur ^ 1, k0 + 64);           // fill ping while pong computes
        }

        // ---- A operands: ISA 16-bit A layout == contiguous 16B runs ----
        const bf16* Ab = As + cur * ABUF + (wave * 16 + ncol) * AST2;
        v16bf a0, a1;
        {
            v8bf lo = *(const v8bf*)(Ab + half * 8);
            v8bf hi = *(const v8bf*)(Ab + 16 + half * 8);
#pragma unroll
            for (int e = 0; e < 8; ++e) { a0[e] = lo[e]; a0[e + 8] = hi[e]; }
            lo = *(const v8bf*)(Ab + 32 + half * 8);
            hi = *(const v8bf*)(Ab + 48 + half * 8);
#pragma unroll
            for (int e = 0; e < 8; ++e) { a1[e] = lo[e]; a1[e + 8] = hi[e]; }
        }

        // ---- 8 WMMA issues per wave per slab (4 cout tiles x 2 K-halves) ----
#pragma unroll
        for (int ni = 0; ni < 4; ++ni) {
            const bf16* Bb = Bs + cur * ABUF + (ni * 16 + ncol) * AST2;
            v16bf b;
            {
                v8bf lo = *(const v8bf*)(Bb + half * 8);
                v8bf hi = *(const v8bf*)(Bb + 16 + half * 8);
#pragma unroll
                for (int e = 0; e < 8; ++e) { b[e] = lo[e]; b[e + 8] = hi[e]; }
            }
            // EXEC all-ones here (uniform control flow) as WMMA requires.
            acc[ni] = __builtin_amdgcn_wmma_f32_16x16x32_bf16(
                          false, a0, false, b, (short)0, acc[ni], false, false);
            {
                v8bf lo = *(const v8bf*)(Bb + 32 + half * 8);
                v8bf hi = *(const v8bf*)(Bb + 48 + half * 8);
#pragma unroll
                for (int e = 0; e < 8; ++e) { b[e] = lo[e]; b[e + 8] = hi[e]; }
            }
            acc[ni] = __builtin_amdgcn_wmma_f32_16x16x32_bf16(
                          false, a1, false, b, (short)0, acc[ni], false, false);
        }
    }

    // ---- epilogue: LDS transpose -> coalesced pixel-contiguous stores ----
    __syncthreads();
#pragma unroll
    for (int ni = 0; ni < 4; ++ni)
#pragma unroll
        for (int v = 0; v < 8; ++v) {
            int m = v + half * 8;                       // pixel within wave tile
            Cs[(ni * 16 + ncol) * 65 + wave * 16 + m] = acc[ni][v];
        }
    __syncthreads();

    for (int i = tid; i < 64 * 64; i += 128) {
        int m = i & 63, cl = i >> 6;
        int p = pbase + m, cout = cbase + cl;
        if (p < HWo && cout < Cout) {
            float o = Cs[cl * 65 + m] + (Bias ? Bias[cout] : 0.f);
            if (act == 1)      o = (o >= 0.f) ? o : 0.2f * o;
            else if (act == 2) o = tanhf(o);
            if (wv && (p & (Wo - 1)) >= wv[n]) o = 0.f;
            Y[((long)n * Cout + cout) * HWo + p] = (bf16)o;
        }
    }
}

// ---------------------------------------------------------------------------
// Elementwise / reduction helpers (grid-stride, bf16 tensors)
// ---------------------------------------------------------------------------
__global__ void k_cvt(const float* __restrict__ s, bf16* __restrict__ d, long n) {
    for (long i = blockIdx.x * 256L + threadIdx.x; i < n; i += gridDim.x * 256L)
        d[i] = (bf16)s[i];
}
__global__ void k_bf2f(const bf16* __restrict__ s, float* __restrict__ d, long n) {
    for (long i = blockIdx.x * 256L + threadIdx.x; i < n; i += gridDim.x * 256L)
        d[i] = (float)s[i];
}
__global__ void k_zero_f32(float* __restrict__ p, long n) {
    for (long i = blockIdx.x * 256L + threadIdx.x; i < n; i += gridDim.x * 256L)
        p[i] = 0.f;
}
__global__ void k_add_scale(const bf16* __restrict__ a, const bf16* __restrict__ b,
                            bf16* __restrict__ y, float s, long n) {
    for (long i = blockIdx.x * 256L + threadIdx.x; i < n; i += gridDim.x * 256L)
        y[i] = (bf16)(((float)a[i] + (float)b[i]) * s);
}
__global__ void k_add_bf_f32(const bf16* __restrict__ a, const float* __restrict__ b,
                             bf16* __restrict__ y, long n) {
    for (long i = blockIdx.x * 256L + threadIdx.x; i < n; i += gridDim.x * 256L)
        y[i] = (bf16)((float)a[i] + b[i]);
}

// bilinear x2 upsample, half-pixel centers (jax.image.resize 'bilinear')
__global__ void k_up2(const bf16* __restrict__ x, bf16* __restrict__ y,
                      int N, int C, int H, int W, long total) {
    int H2 = 2 * H, W2 = 2 * W;
    for (long i = blockIdx.x * 256L + threadIdx.x; i < total; i += gridDim.x * 256L) {
        int ox = (int)(i % W2); long t = i / W2;
        int oy = (int)(t % H2); long nc = t / H2;
        float sy = (oy + 0.5f) * 0.5f - 0.5f;
        float sx = (ox + 0.5f) * 0.5f - 0.5f;
        int y0 = (int)floorf(sy); float fy = sy - (float)y0;
        int x0 = (int)floorf(sx); float fx = sx - (float)x0;
        int y0c = iclampi(y0, 0, H - 1), y1c = iclampi(y0 + 1, 0, H - 1);
        int x0c = iclampi(x0, 0, W - 1), x1c = iclampi(x0 + 1, 0, W - 1);
        const bf16* p = x + nc * (long)H * W;
        float v00 = (float)p[(long)y0c * W + x0c], v01 = (float)p[(long)y0c * W + x1c];
        float v10 = (float)p[(long)y1c * W + x0c], v11 = (float)p[(long)y1c * W + x1c];
        float v = v00 * (1.f - fy) * (1.f - fx) + v01 * (1.f - fy) * fx
                + v10 * fy * (1.f - fx) + v11 * fy * fx;
        y[i] = (bf16)v;
    }
}

__global__ void k_concat(const bf16* __restrict__ a, int Ca,
                         const bf16* __restrict__ b, int Cb,
                         bf16* __restrict__ y, int H, int W, long total) {
    int Ct = Ca + Cb;
    for (long i = blockIdx.x * 256L + threadIdx.x; i < total; i += gridDim.x * 256L) {
        int w = (int)(i % W); long t = i / W;
        int h = (int)(t % H); t /= H;
        int c = (int)(t % Ct); long n = t / Ct;
        float v = (c < Ca) ? (float)a[((n * Ca + c) * (long)H + h) * W + w]
                           : (float)b[((n * Cb + (c - Ca)) * (long)H + h) * W + w];
        y[i] = (bf16)v;
    }
}

// window geometry from character centers
__global__ void k_locs(const int* __restrict__ locs, int* x1, int* y1, int* wv,
                       int n, int divv, int width, int W) {
    int i = threadIdx.x;
    if (i < n) {
        int cen = locs[i] / divv;
        int a = (cen < width) ? 0 : (cen - width);
        int b = cen + width; if (b > W) b = W;
        x1[i] = a;
        y1[i] = (cen < width) ? (width - cen) : 0;
        wv[i] = b - a;
    }
}

// masked window gather from the squeeze features: [K,C,H,ww] <- [B,C,H,W]
__global__ void k_gather_lq(const bf16* __restrict__ sq, bf16* __restrict__ out,
                            const int* __restrict__ x1, const int* __restrict__ wv,
                            int nc, int C, int H, int W, int WW, long total) {
    for (long i = blockIdx.x * 256L + threadIdx.x; i < total; i += gridDim.x * 256L) {
        int j = (int)(i % WW); long t = i / WW;
        int h = (int)(t % H); t /= H;
        int c = (int)(t % C); int k = (int)(t / C);
        int b = k / nc;
        float v = 0.f;
        if (j < wv[k]) {
            int col = x1[k] + j;
            if (col < W)
                v = (float)sq[(((long)b * C + c) * H + h) * W + col];
        }
        out[i] = (bf16)v;
    }
}

// masked window gather from the priors: [K,C,H,ww] <- [K,C,H,ww] shifted by y1
__global__ void k_gather_pr(const bf16* __restrict__ pri, bf16* __restrict__ out,
                            const int* __restrict__ y1, const int* __restrict__ wv,
                            int C, int H, int WW, long total) {
    for (long i = blockIdx.x * 256L + threadIdx.x; i < total; i += gridDim.x * 256L) {
        int j = (int)(i % WW); long t = i / WW;
        int h = (int)(t % H); t /= H;
        int c = (int)(t % C); int k = (int)(t / C);
        float v = 0.f;
        if (j < wv[k]) {
            int col = y1[k] + j;
            if (col < WW)
                v = (float)pri[(((long)k * C + c) * H + h) * WW + col];
        }
        out[i] = (bf16)v;
    }
}

// masked per-(k,c) mean / std (var with N-1), one block per (k*C + c)
__global__ void k_stats(const bf16* __restrict__ x, const int* __restrict__ wv,
                        float* __restrict__ mean, float* __restrict__ stdv,
                        int C, int H, int WW) {
    __shared__ float red[256];
    int kc = blockIdx.x;
    int k  = kc / C;
    int w  = wv[k];
    long base = (long)kc * H * WW;
    int tot = H * WW;

    float s = 0.f;
    for (int i = threadIdx.x; i < tot; i += 256) {
        int j = i % WW;
        if (j < w) s += (float)x[base + i];
    }
    red[threadIdx.x] = s; __syncthreads();
    for (int o = 128; o > 0; o >>= 1) {
        if (threadIdx.x < o) red[threadIdx.x] += red[threadIdx.x + o];
        __syncthreads();
    }
    float cnt = fmaxf((float)(H * w), 1.f);
    float m = red[0] / cnt;
    __syncthreads();

    float s2 = 0.f;
    for (int i = threadIdx.x; i < tot; i += 256) {
        int j = i % WW;
        if (j < w) { float d = (float)x[base + i] - m; s2 += d * d; }
    }
    red[threadIdx.x] = s2; __syncthreads();
    for (int o = 128; o > 0; o >>= 1) {
        if (threadIdx.x < o) red[threadIdx.x] += red[threadIdx.x + o];
        __syncthreads();
    }
    if (threadIdx.x == 0) {
        mean[kc] = m;
        stdv[kc] = sqrtf(red[0] / fmaxf(cnt - 1.f, 1.f) + 1e-5f);
    }
}

// adain: ((pr - mc)/sc)*ss + ms, masked
__global__ void k_adain(const bf16* __restrict__ pr, bf16* __restrict__ out,
                        const float* __restrict__ mc, const float* __restrict__ sc,
                        const float* __restrict__ ms, const float* __restrict__ ss,
                        const int* __restrict__ wv, int C, int H, int WW, long total) {
    for (long i = blockIdx.x * 256L + threadIdx.x; i < total; i += gridDim.x * 256L) {
        int j = (int)(i % WW);
        long kc = i / ((long)H * WW);
        int k = (int)(kc / C);
        float v = 0.f;
        if (j < wv[k])
            v = (((float)pr[i] - mc[kc]) / sc[kc]) * ss[kc] + ms[kc];
        out[i] = (bf16)v;
    }
}

// contrib = (lq*scale + shift) masked, scatter-add into [B,C,H,W] f32
__global__ void k_scatter(const bf16* __restrict__ lqm, const bf16* __restrict__ scl,
                          const bf16* __restrict__ shf, float* __restrict__ acc,
                          const int* __restrict__ x1, const int* __restrict__ wv,
                          int nc, int C, int H, int W, int WW, long total) {
    for (long i = blockIdx.x * 256L + threadIdx.x; i < total; i += gridDim.x * 256L) {
        int j = (int)(i % WW); long t = i / WW;
        int h = (int)(t % H); t /= H;
        int c = (int)(t % C); int k = (int)(t / C);
        int b = k / nc;
        if (j < wv[k]) {
            int col = x1[k] + j;
            if (col < W) {
                float v = (float)lqm[i] * (float)scl[i] + (float)shf[i];
                atomicAdd(&acc[(((long)b * C + c) * H + h) * W + col], v);
            }
        }
    }
}

// ---------------------------------------------------------------------------
static unsigned gs(long n) {
    long b = (n + 255) / 256;
    if (b < 1) b = 1;
    if (b > 1048576) b = 1048576;
    return (unsigned)b;
}

// Params flatten order (jax dict tree: sorted keys, leaves b before w,
// res blocks as c1,c2,sk). d_in[0..3] = lq, priors64, priors32, locs; then
// pairs (b,w) starting at index 4. Pair ids:
enum {
    P_body16a = 0, P_body16b, P_body32a, P_body32b,
    P_fin_c1, P_fin_c2, P_fin_c3,
    P_finres_c1, P_finres_c2, P_finres_sk,
    P_first16, P_first32, P_first8a, P_first8b,
    P_fuse32_c1, P_fuse32_c2, P_fuse32_sk,
    P_fuse64_c1, P_fuse64_c2, P_fuse64_sk,
    P_s32a, P_s32b, P_s64a, P_s64b,
    P_sh32a, P_sh32b, P_sh64a, P_sh64b,
    P_to256a, P_to256b, P_up_c1, P_up_c2,
    P_upres_c1, P_upres_c2, P_upres_sk
};

extern "C" void kernel_launch(void* const* d_in, const int* in_sizes, int n_in,
                              void* d_out, int out_size, void* d_ws, size_t ws_size,
                              hipStream_t stream)
{
    (void)in_sizes; (void)n_in; (void)out_size; (void)ws_size;

    const float* lq   = (const float*)d_in[0];
    const float* p64  = (const float*)d_in[1];
    const float* p32  = (const float*)d_in[2];
    const int*   locs = (const int*)d_in[3];

    const int B = 2, NC = 8, KC = 16;
    const float RS2 = 0.70710678f;

    // deterministic bump allocator over d_ws
    size_t cur = 0;
    auto alloc = [&](size_t bytes) -> void* {
        void* p = (char*)d_ws + cur;
        cur = (cur + bytes + 255) & ~(size_t)255;
        return p;
    };

    bf16* LQBF  = (bf16*)alloc((size_t)B * 3 * 32 * 256 * 2);
    bf16* F32A  = (bf16*)alloc((size_t)B * 64 * 32 * 256 * 2);
    bf16* F16   = (bf16*)alloc((size_t)B * 128 * 16 * 128 * 2);
    bf16* F8    = (bf16*)alloc((size_t)B * 256 * 8 * 64 * 2);
    bf16* SQ16  = (bf16*)alloc((size_t)B * 256 * 16 * 128 * 2);
    bf16* SQ32  = (bf16*)alloc((size_t)B * 256 * 32 * 256 * 2);
    bf16* PR32  = (bf16*)alloc((size_t)KC * 256 * 32 * 32 * 2);
    bf16* SQ32O = (bf16*)alloc((size_t)B * 256 * 32 * 256 * 2);
    bf16* F64   = (bf16*)alloc((size_t)B * 256 * 64 * 512 * 2);
    bf16* SQ64  = (bf16*)alloc((size_t)B * 256 * 64 * 512 * 2);
    bf16* UP    = (bf16*)alloc((size_t)33554432 * 2);  // up to 128ch@128x1024xB2
    bf16* CATB  = (bf16*)alloc((size_t)B * 320 * 32 * 256 * 2);
    bf16* TA    = (bf16*)alloc((size_t)16777216 * 2);
    bf16* TB    = (bf16*)alloc((size_t)16777216 * 2);
    bf16* TC    = (bf16*)alloc((size_t)16777216 * 2);
    bf16* LQM   = (bf16*)alloc((size_t)16777216 * 2);
    bf16* PRM   = (bf16*)alloc((size_t)16777216 * 2);
    bf16* AD    = (bf16*)alloc((size_t)16777216 * 2);
    bf16* CATW  = (bf16*)alloc((size_t)33554432 * 2);  // 512ch windows / priors32 in
    bf16* FUSE  = (bf16*)alloc((size_t)16777216 * 2);
    bf16* SCL   = (bf16*)alloc((size_t)16777216 * 2);
    bf16* SHF   = (bf16*)alloc((size_t)16777216 * 2);
    float* ACC  = (float*)alloc((size_t)16777216 * 4);
    bf16* WSCR  = (bf16*)alloc((size_t)1179648 * 2);   // largest weight 256*512*9
    float* MC   = (float*)alloc(4096 * 4);
    float* SC   = (float*)alloc(4096 * 4);
    float* MS   = (float*)alloc(4096 * 4);
    float* SS   = (float*)alloc(4096 * 4);
    int* X1     = (int*)alloc(64);
    int* Y1     = (int*)alloc(64);
    int* WV     = (int*)alloc(64);
    bf16* OB    = (bf16*)alloc((size_t)B * 3 * 128 * 1024 * 2);

    auto conv = [&](int pair, const bf16* X, bf16* Y, int N, int Cin, int H, int W,
                    int Cout, int ks, int stride, int act, const int* wvp) {
        const float* Wf = (const float*)d_in[5 + 2 * pair];
        const float* Bf = (const float*)d_in[4 + 2 * pair];
        long wn = (long)Cout * Cin * ks * ks;
        k_cvt<<<gs(wn), 256, 0, stream>>>(Wf, WSCR, wn);
        int Ho = H / stride, Wo = W / stride;
        int woShift = __builtin_ctz(Wo);   // all layer widths are powers of two
        dim3 g((unsigned)((Ho * Wo + 63) / 64), (unsigned)((Cout + 63) / 64), (unsigned)N);
        if (ks == 3 && stride == 1)
            k_conv_wmma<3, 1><<<g, 128, 0, stream>>>(X, WSCR, Bf, Y, N, Cin, H, W,
                                                     Cout, Ho, woShift, act, wvp);
        else if (ks == 3 && stride == 2)
            k_conv_wmma<3, 2><<<g, 128, 0, stream>>>(X, WSCR, Bf, Y, N, Cin, H, W,
                                                     Cout, Ho, woShift, act, wvp);
        else
            k_conv_wmma<1, 1><<<g, 128, 0, stream>>>(X, WSCR, Bf, Y, N, Cin, H, W,
                                                     Cout, Ho, woShift, act, wvp);
    };

    long n;
    // ---------------- trunk ----------------
    n = (long)B * 3 * 32 * 256;       k_cvt<<<gs(n), 256, 0, stream>>>(lq, LQBF, n);
    conv(P_first32, LQBF, F32A, B, 3, 32, 256, 64, 3, 1, 1, nullptr);
    conv(P_first16, F32A, F16,  B, 64, 32, 256, 128, 3, 2, 1, nullptr);
    conv(P_first8a, F16,  TA,   B, 128, 16, 128, 256, 3, 2, 1, nullptr);
    conv(P_first8b, TA,   F8,   B, 256, 8, 64,   256, 3, 1, 0, nullptr);

    n = (long)B * 256 * 16 * 128;     k_up2<<<gs(n), 256, 0, stream>>>(F8, UP, B, 256, 8, 64, n);
    n = (long)B * 384 * 16 * 128;     k_concat<<<gs(n), 256, 0, stream>>>(UP, 256, F16, 128, CATB, 16, 128, n);
    conv(P_body16a, CATB, TA,   B, 384, 16, 128, 256, 3, 1, 1, nullptr);
    conv(P_body16b, TA,   SQ16, B, 256, 16, 128, 256, 3, 1, 0, nullptr);

    n = (long)B * 256 * 32 * 256;     k_up2<<<gs(n), 256, 0, stream>>>(SQ16, UP, B, 256, 16, 128, n);
    n = (long)B * 320 * 32 * 256;     k_concat<<<gs(n), 256, 0, stream>>>(UP, 256, F32A, 64, CATB, 32, 256, n);
    conv(P_body32a, CATB, TA,   B, 320, 32, 256, 256, 3, 1, 1, nullptr);
    conv(P_body32b, TA,   SQ32, B, 256, 32, 256, 256, 3, 1, 0, nullptr);

    // ---------------- prior branch @32 ----------------
    n = (long)KC * 512 * 32 * 32;     k_cvt<<<gs(n), 256, 0, stream>>>(p32, CATW, n);
    conv(P_to256a, CATW, TA,   KC, 512, 32, 32, 256, 3, 1, 1, nullptr);
    conv(P_to256b, TA,   PR32, KC, 256, 32, 32, 256, 3, 1, 0, nullptr);

    k_locs<<<1, 32, 0, stream>>>(locs, X1, Y1, WV, KC, 4, 16, 256);
    n = (long)KC * 256 * 32 * 32;
    k_gather_lq<<<gs(n), 256, 0, stream>>>(SQ32, LQM, X1, WV, NC, 256, 32, 256, 32, n);
    k_gather_pr<<<gs(n), 256, 0, stream>>>(PR32, PRM, Y1, WV, 256, 32, 32, n);
    k_stats<<<KC * 256, 256, 0, stream>>>(PRM, WV, MC, SC, 256, 32, 32);
    k_stats<<<KC * 256, 256, 0, stream>>>(LQM, WV, MS, SS, 256, 32, 32);
    k_adain<<<gs(n), 256, 0, stream>>>(PRM, AD, MC, SC, MS, SS, WV, 256, 32, 32, n);
    {
        long n2 = (long)KC * 512 * 32 * 32;
        k_concat<<<gs(n2), 256, 0, stream>>>(AD, 256, LQM, 256, CATW, 32, 32, n2);
    }
    conv(P_fuse32_c1, CATW, TA, KC, 512, 32, 32, 256, 3, 1, 1, WV);
    conv(P_fuse32_c2, TA,   TB, KC, 256, 32, 32, 256, 3, 1, 0, WV);
    conv(P_fuse32_sk, CATW, TC, KC, 512, 32, 32, 256, 1, 1, 0, WV);
    k_add_scale<<<gs(n), 256, 0, stream>>>(TB, TC, FUSE, RS2, n);
    conv(P_s32a,  FUSE, TA,  KC, 256, 32, 32, 256, 3, 1, 1, WV);
    conv(P_s32b,  TA,   SCL, KC, 256, 32, 32, 256, 3, 1, 0, WV);
    conv(P_sh32a, FUSE, TA,  KC, 256, 32, 32, 256, 3, 1, 1, WV);
    conv(P_sh32b, TA,   SHF, KC, 256, 32, 32, 256, 3, 1, 0, WV);
    {
        long na = (long)B * 256 * 32 * 256;
        k_zero_f32<<<gs(na), 256, 0, stream>>>(ACC, na);
        k_scatter<<<gs(n), 256, 0, stream>>>(LQM, SCL, SHF, ACC, X1, WV, NC, 256, 32, 256, 32, n);
        k_add_bf_f32<<<gs(na), 256, 0, stream>>>(SQ32, ACC, SQ32O, na);
    }

    // ---------------- upsample trunk to 64 ----------------
    n = (long)B * 256 * 64 * 512;     k_up2<<<gs(n), 256, 0, stream>>>(SQ32O, UP, B, 256, 32, 256, n);
    conv(P_up_c1,    UP, TA, B, 256, 64, 512, 256, 3, 1, 1, nullptr);
    conv(P_upres_c1, TA, TB, B, 256, 64, 512, 256, 3, 1, 1, nullptr);
    conv(P_upres_c2, TB, TC, B, 256, 64, 512, 256, 3, 1, 0, nullptr);
    conv(P_upres_sk, TA, FUSE, B, 256, 64, 512, 256, 1, 1, 0, nullptr);
    k_add_scale<<<gs(n), 256, 0, stream>>>(TC, FUSE, TB, RS2, n);
    conv(P_up_c2, TB, F64, B, 256, 64, 512, 256, 3, 1, 0, nullptr);

    // ---------------- prior branch @64 ----------------
    n = (long)KC * 256 * 64 * 64;     k_cvt<<<gs(n), 256, 0, stream>>>(p64, UP, n);
    k_locs<<<1, 32, 0, stream>>>(locs, X1, Y1, WV, KC, 2, 32, 512);
    k_gather_lq<<<gs(n), 256, 0, stream>>>(F64, LQM, X1, WV, NC, 256, 64, 512, 64, n);
    k_gather_pr<<<gs(n), 256, 0, stream>>>(UP, PRM, Y1, WV, 256, 64, 64, n);
    k_stats<<<KC * 256, 256, 0, stream>>>(PRM, WV, MC, SC, 256, 64, 64);
    k_stats<<<KC * 256, 256, 0, stream>>>(LQM, WV, MS, SS, 256, 64, 64);
    k_adain<<<gs(n), 256, 0, stream>>>(PRM, AD, MC, SC, MS, SS, WV, 256, 64, 64, n);
    {
        long n2 = (long)KC * 512 * 64 * 64;
        k_concat<<<gs(n2), 256, 0, stream>>>(AD, 256, LQM, 256, CATW, 64, 64, n2);
    }
    conv(P_fuse64_c1, CATW, TA, KC, 512, 64, 64, 256, 3, 1, 1, WV);
    conv(P_fuse64_c2, TA,   TB, KC, 256, 64, 64, 256, 3, 1, 0, WV);
    conv(P_fuse64_sk, CATW, TC, KC, 512, 64, 64, 256, 1, 1, 0, WV);
    k_add_scale<<<gs(n), 256, 0, stream>>>(TB, TC, FUSE, RS2, n);
    conv(P_s64a,  FUSE, TA,  KC, 256, 64, 64, 256, 3, 1, 1, WV);
    conv(P_s64b,  TA,   SCL, KC, 256, 64, 64, 256, 3, 1, 0, WV);
    conv(P_sh64a, FUSE, TA,  KC, 256, 64, 64, 256, 3, 1, 1, WV);
    conv(P_sh64b, TA,   SHF, KC, 256, 64, 64, 256, 3, 1, 0, WV);
    {
        long na = (long)B * 256 * 64 * 512;
        k_zero_f32<<<gs(na), 256, 0, stream>>>(ACC, na);
        k_scatter<<<gs(n), 256, 0, stream>>>(LQM, SCL, SHF, ACC, X1, WV, NC, 256, 64, 512, 64, n);
        k_add_bf_f32<<<gs(na), 256, 0, stream>>>(F64, ACC, SQ64, na);
    }

    // ---------------- head ----------------
    conv(P_fin_c1, SQ64, TA, B, 256, 64, 512, 128, 3, 1, 1, nullptr);
    n = (long)B * 128 * 128 * 1024;   k_up2<<<gs(n), 256, 0, stream>>>(TA, UP, B, 128, 64, 512, n);
    conv(P_fin_c2,    UP, TB, B, 128, 128, 1024, 64, 3, 1, 1, nullptr);
    conv(P_finres_c1, TB, TC, B, 64, 128, 1024, 64, 3, 1, 1, nullptr);
    conv(P_finres_c2, TC, FUSE, B, 64, 128, 1024, 64, 3, 1, 0, nullptr);
    conv(P_finres_sk, TB, SCL,  B, 64, 128, 1024, 64, 1, 1, 0, nullptr);
    n = (long)B * 64 * 128 * 1024;
    k_add_scale<<<gs(n), 256, 0, stream>>>(FUSE, SCL, TC, RS2, n);
    conv(P_fin_c3, TC, OB, B, 64, 128, 1024, 3, 3, 1, 2, nullptr);
    n = (long)B * 3 * 128 * 1024;
    k_bf2f<<<gs(n), 256, 0, stream>>>(OB, (float*)d_out, n);
}
